// GATActor_20126216749448
// MI455X (gfx1250) — compile-verified, matrix operand order checked by reading, not run on hardware
//
#include <hip/hip_runtime.h>
#include <hip/hip_bf16.h>
#include <stdint.h>

// ---------------- problem constants ----------------
#define NUM_GRAPHS 2048
#define NODES_PER_GRAPH 32
#define NUM_AGENTS 5
#define DEG 8
#define HEADS 8
#define HID 32
#define HC 256                 // HEADS*HID
#define ACT_DIM 3
#define NNODES (NUM_GRAPHS*NODES_PER_GRAPH)   // 65536
#define NEDGES (NNODES*DEG)                   // 524288
#define NROWS (NUM_GRAPHS*NUM_AGENTS)         // 10240
#define FC1_IN 512
#define FC1_OUT 128

// ---------------- vector types (trivial, union-safe) ----------------
typedef __attribute__((ext_vector_type(16))) __bf16 v16bf;
typedef __attribute__((ext_vector_type(8)))  float  v8f;
typedef __attribute__((ext_vector_type(2)))  float  v2f;
typedef __attribute__((ext_vector_type(4)))  unsigned int u32x4;
typedef __attribute__((ext_vector_type(4)))  float  f32x4;

union FragBF { v16bf v; u32x4 u[2]; };

__device__ __forceinline__ __bf16 f2bf(float f) {
  unsigned u = __float_as_uint(f);
  unsigned r = (u + 0x7FFFu + ((u >> 16) & 1u)) >> 16;   // round-to-nearest-even
  unsigned short s = (unsigned short)r;
  return __builtin_bit_cast(__bf16, s);
}

// ---------------- weight prep: transpose + bf16 convert ----------------
__global__ void prep_kernel(const float* __restrict__ lin1, const float* __restrict__ lin2,
                            const float* __restrict__ lin3, const float* __restrict__ fc1w,
                            float* __restrict__ W1t, __bf16* __restrict__ W2t,
                            __bf16* __restrict__ W3t, __bf16* __restrict__ fc1t) {
  int t = blockIdx.x * blockDim.x + threadIdx.x;   // 0..65535
  { int n = t >> 8, k = t & 255;
    W2t[t] = f2bf(lin2[k * HC + n]);
    W3t[t] = f2bf(lin3[k * HC + n]); }
  { int n = t >> 9, k = t & 511;
    fc1t[t] = f2bf(fc1w[k * FC1_OUT + n]); }
  if (t < HC * 4) { int n = t >> 2, k = t & 3; W1t[t] = lin1[k * HC + n]; }
}

// ---------------- conv1 lin: fp32 WMMA, K = 4 exactly ----------------
__global__ void conv1_gemm_kernel(const float* __restrict__ X, const float* __restrict__ W1t,
                                  float* __restrict__ hlin) {
  int wid = (blockIdx.x * blockDim.x + threadIdx.x) >> 5;
  int l   = threadIdx.x & 31;
  int tm = wid >> 4, tn = wid & 15;               // 4096 x 16 tiles
  int m  = tm * 16 + (l & 15);
  int n  = tn * 16 + (l & 15);
  int k0 = (l >> 4) * 2;                          // 32-bit A 16x4: half-lanes split K
  v2f a, b;
  a[0] = X[m * 4 + k0];     a[1] = X[m * 4 + k0 + 1];
  b[0] = W1t[n * 4 + k0];   b[1] = W1t[n * 4 + k0 + 1];
  v8f c = {};
  c = __builtin_amdgcn_wmma_f32_16x16x4_f32(false, a, false, b, (short)0, c, false, false);
  int rb = tm * 16 + ((l >> 4) << 3);
  for (int r = 0; r < 8; ++r)
    hlin[(size_t)(rb + r) * HC + n] = c[r];
}

// ---------------- bf16 WMMA GEMM, 16x64 tile per wave ----------------
// C[M,N] = A[M,K] x Bt[N,K]^T ; K compile-time; 4 N-tiles per wave so the A
// fragment feeds 4 back-to-back v_wmma ops (A traffic /4, latency hidden).
template<int K>
__global__ void gemm_bf16_kernel(const __bf16* __restrict__ A, const __bf16* __restrict__ Bt,
                                 const float* __restrict__ bias, float* __restrict__ C,
                                 int M, int Nn, int relu) {
  constexpr int TN = 4;
  int wid = (blockIdx.x * blockDim.x + threadIdx.x) >> 5;
  int l   = threadIdx.x & 31;
  int ngrp = Nn / (16 * TN);
  int tm = wid / ngrp, tg = wid - tm * ngrp;
  if (tm * 16 >= M) return;                        // wave-uniform
  int m  = tm * 16 + (l & 15);
  int n0 = tg * 16 * TN + (l & 15);
  int kb = (l >> 4) * 8;                           // 16-bit A layout: half-lanes split K

  const u32x4* arow = (const u32x4*)(A + (size_t)m * K + kb);
  const u32x4* brow[TN];
  #pragma unroll
  for (int t = 0; t < TN; ++t)
    brow[t] = (const u32x4*)(Bt + (size_t)(n0 + 16 * t) * K + kb);

  v8f acc[TN];
  #pragma unroll
  for (int t = 0; t < TN; ++t) acc[t] = (v8f){};

  #pragma unroll 4
  for (int kc = 0; kc < K; kc += 32) {
    int ki = kc >> 3;                              // u32x4 index (8 bf16 each)
    FragBF fa;
    fa.u[0] = arow[ki]; fa.u[1] = arow[ki + 2];    // K = kb..kb+7, 16+kb..16+kb+7
    #pragma unroll
    for (int t = 0; t < TN; ++t) {
      FragBF fb;
      fb.u[0] = brow[t][ki]; fb.u[1] = brow[t][ki + 2];
      acc[t] = __builtin_amdgcn_wmma_f32_16x16x32_bf16(false, fa.v, false, fb.v,
                                                       (short)0, acc[t], false, false);
    }
  }

  int rb = tm * 16 + ((l >> 4) << 3);
  #pragma unroll
  for (int t = 0; t < TN; ++t) {
    int n = n0 + 16 * t;
    float bv = bias ? bias[n] : 0.f;
    #pragma unroll
    for (int r = 0; r < 8; ++r) {
      float v = acc[t][r] + bv;
      if (relu) v = fmaxf(v, 0.f);
      C[(size_t)(rb + r) * Nn + n] = v;
    }
  }
}

// ---------------- per-node attention coefficients ----------------
__global__ void attn_kernel(const float* __restrict__ hlin, const float* __restrict__ att_src,
                            const float* __restrict__ att_dst,
                            float* __restrict__ asrc, float* __restrict__ adst) {
  int id = blockIdx.x * blockDim.x + threadIdx.x;  // < N*8
  int node = id >> 3, head = id & 7;
  const float* hp = hlin + (size_t)node * HC + head * HID;
  const float* ps = att_src + head * HID;
  const float* pd = att_dst + head * HID;
  float a = 0.f, b = 0.f;
  for (int c = 0; c < HID; ++c) { float h = hp[c]; a += h * ps[c]; b += h * pd[c]; }
  asrc[id] = a; adst[id] = b;
}

// ---------------- attention softmax + neighbor aggregation (one graph / block) ----------------
// Feature tile staged with gfx1250 async DMA loads (global -> LDS, no VGPR
// round-trip, tracked by ASYNCcnt), then per-node softmax over 8(+1) contiguous edges.
__global__ void aggregate_kernel(const float* __restrict__ hlin,
                                 const float* __restrict__ asrc, const float* __restrict__ adst,
                                 const int* __restrict__ src_idx, const float* __restrict__ ea,
                                 const float* __restrict__ lin_edge, const float* __restrict__ att_edge,
                                 const float* __restrict__ bias,
                                 float* __restrict__ hout, __bf16* __restrict__ hout_bf,
                                 int with_self_loop) {
  __shared__ __align__(16) float hl[NODES_PER_GRAPH * HC];   // 32KB
  __shared__ float as_l[NODES_PER_GRAPH * HEADS];
  __shared__ float ad_l[NODES_PER_GRAPH * HEADS];
  __shared__ float eav[NODES_PER_GRAPH * DEG];
  __shared__ int   srcs[NODES_PER_GRAPH * DEG];
  __shared__ float ke[HEADS];

  int g = blockIdx.x, t = threadIdx.x;

  // async DMA the 32x256 f32 feature tile into LDS (8 x b128 per thread)
  #pragma unroll
  for (int i = 0; i < 8; ++i) {
    int v = i * 256 + t;
    unsigned lds_addr = (unsigned)(uintptr_t)(hl + v * 4);       // low 32b = LDS byte offset
    const f32x4* gptr = ((const f32x4*)hlin) + (size_t)g * 2048 + v;
    asm volatile("global_load_async_to_lds_b128 %0, %1, off"
                 :: "v"(lds_addr), "v"(gptr) : "memory");
  }
  as_l[t] = asrc[g * 256 + t];
  ad_l[t] = adst[g * 256 + t];
  srcs[t] = src_idx[g * 256 + t] - g * NODES_PER_GRAPH;
  eav[t]  = ea[g * 256 + t];
  if (t < HEADS) {
    float s = 0.f;
    for (int c = 0; c < HID; ++c) s += lin_edge[t * HID + c] * att_edge[t * HID + c];
    ke[t] = s;
  }
  asm volatile("s_wait_asynccnt 0" ::: "memory");
  __syncthreads();

  int w = t >> 5, l = t & 31;
  int head = l >> 2;            // 8 channels per lane -> one head per lane
  int c0 = l * 8;
  for (int j = 0; j < 4; ++j) {
    int ln = w * 4 + j;
    int gn = g * NODES_PER_GRAPH + ln;
    float adv = ad_l[ln * HEADS + head];
    float ap[9];
    float easum = 0.f;
    for (int e = 0; e < DEG; ++e) {
      int s = srcs[ln * DEG + e];
      float v = as_l[s * HEADS + head] + adv + eav[ln * DEG + e] * ke[head];
      ap[e] = v > 0.f ? v : 0.2f * v;              // leaky_relu(0.2)
      easum += eav[ln * DEG + e];
    }
    int ne = DEG;
    if (with_self_loop) {
      float la = easum * (1.f / DEG);              // fill_value='mean'
      float v = as_l[ln * HEADS + head] + adv + la * ke[head];
      ap[DEG] = v > 0.f ? v : 0.2f * v;
      ne = DEG + 1;
    }
    float mx = ap[0];
    for (int e = 1; e < ne; ++e) mx = fmaxf(mx, ap[e]);
    float den = 0.f;
    for (int e = 0; e < ne; ++e) { ap[e] = __expf(ap[e] - mx); den += ap[e]; }
    float inv = 1.f / (den + 1e-16f);
    float acc[8];
    for (int i = 0; i < 8; ++i) acc[i] = 0.f;
    for (int e = 0; e < ne; ++e) {
      int s = (e < DEG) ? srcs[ln * DEG + e] : ln;
      float wgt = ap[e] * inv;
      const float* hp = hl + s * HC + c0;
      for (int i = 0; i < 8; ++i) acc[i] += wgt * hp[i];
    }
    size_t row = (size_t)gn * HC + c0;
    for (int i = 0; i < 8; ++i) {
      float v = fmaxf(acc[i] + bias[c0 + i], 0.f); // +bias, relu (all 3 layers)
      hout[row + i] = v;
      hout_bf[row + i] = f2bf(v);
    }
  }
}

// ---------------- mean pool + build fc1 input (bf16) ----------------
__global__ void pool_comb_kernel(const float* __restrict__ h3, __bf16* __restrict__ comb) {
  int g = blockIdx.x, c = threadIdx.x;             // c in [0,256)
  float s = 0.f;
  for (int n = 0; n < NODES_PER_GRAPH; ++n) s += h3[(size_t)(g * NODES_PER_GRAPH + n) * HC + c];
  float gm = s * (1.f / NODES_PER_GRAPH);
  __bf16 gmb = f2bf(gm);
  for (int a = 0; a < NUM_AGENTS; ++a) {
    size_t row = (size_t)(g * NUM_AGENTS + a) * FC1_IN;
    comb[row + c]       = f2bf(h3[(size_t)(g * NODES_PER_GRAPH + a) * HC + c]);
    comb[row + HC + c]  = gmb;
  }
}

// ---------------- fc2 + write mean/std outputs ----------------
__global__ void fc2_kernel(const float* __restrict__ fc1out, const float* __restrict__ w,
                           const float* __restrict__ b, const float* __restrict__ log_std,
                           float* __restrict__ out) {
  int id = blockIdx.x * blockDim.x + threadIdx.x;  // < NROWS*ACT_DIM
  if (id >= NROWS * ACT_DIM) return;
  int row = id / ACT_DIM, j = id - row * ACT_DIM;
  float acc = b[j];
  const float* f = fc1out + (size_t)row * FC1_OUT;
  for (int k = 0; k < FC1_OUT; ++k) acc += f[k] * w[k * ACT_DIM + j];
  out[id] = acc;                                   // mean
  out[NROWS * ACT_DIM + id] = __expf(log_std[j]);  // std (broadcast)
}

// ---------------- launch ----------------
extern "C" void kernel_launch(void* const* d_in, const int* in_sizes, int n_in,
                              void* d_out, int out_size, void* d_ws, size_t ws_size,
                              hipStream_t stream) {
  // inputs per setup_inputs() + sorted-key pytree flattening of params
  const float* x    = (const float*)d_in[0];
  const int*   eidx = (const int*)d_in[1];      // row0 = src (E entries), row1 = dst
  const int*   src  = eidx;
  const float* ea   = (const float*)d_in[2];
  const float* att_dst1  = (const float*)d_in[4];
  const float* att_edge1 = (const float*)d_in[5];
  const float* att_src1  = (const float*)d_in[6];
  const float* bias1     = (const float*)d_in[7];
  const float* lin1      = (const float*)d_in[8];
  const float* lin_edge1 = (const float*)d_in[9];
  const float* att_dst2  = (const float*)d_in[10];
  const float* att_edge2 = (const float*)d_in[11];
  const float* att_src2  = (const float*)d_in[12];
  const float* bias2     = (const float*)d_in[13];
  const float* lin2      = (const float*)d_in[14];
  const float* lin_edge2 = (const float*)d_in[15];
  const float* att_dst3  = (const float*)d_in[16];
  const float* att_edge3 = (const float*)d_in[17];
  const float* att_src3  = (const float*)d_in[18];
  const float* bias3     = (const float*)d_in[19];
  const float* lin3      = (const float*)d_in[20];
  const float* lin_edge3 = (const float*)d_in[21];
  const float* fc1_b     = (const float*)d_in[22];
  const float* fc1_w     = (const float*)d_in[23];
  const float* fc2_b     = (const float*)d_in[24];
  const float* fc2_w     = (const float*)d_in[25];
  const float* log_std   = (const float*)d_in[26];

  // workspace carve
  char* p = (char*)d_ws;
  auto alloc = [&](size_t bytes) { void* r = (void*)p; p += (bytes + 255) & ~(size_t)255; return r; };
  float*  hlin  = (float*) alloc((size_t)NNODES * HC * 4);   // GEMM result (pre-aggregation)
  float*  hbuf  = (float*) alloc((size_t)NNODES * HC * 4);   // layer output f32
  __bf16* hbf   = (__bf16*)alloc((size_t)NNODES * HC * 2);   // layer output bf16 (next GEMM A)
  float*  asrc  = (float*) alloc((size_t)NNODES * HEADS * 4);
  float*  adst  = (float*) alloc((size_t)NNODES * HEADS * 4);
  __bf16* W2t   = (__bf16*)alloc(HC * HC * 2);
  __bf16* W3t   = (__bf16*)alloc(HC * HC * 2);
  __bf16* fc1t  = (__bf16*)alloc(FC1_OUT * FC1_IN * 2);
  float*  W1t   = (float*) alloc(HC * 4 * 4);
  __bf16* comb  = (__bf16*)alloc((size_t)NROWS * FC1_IN * 2);
  float*  fc1o  = (float*) alloc((size_t)NROWS * FC1_OUT * 4);
  float*  out   = (float*)d_out;

  // 0) weight prep
  prep_kernel<<<256, 256, 0, stream>>>(lin1, lin2, lin3, fc1_w, W1t, W2t, W3t, fc1t);

  // ---- conv1 ----
  conv1_gemm_kernel<<<8192, 256, 0, stream>>>(x, W1t, hlin);
  attn_kernel<<<2048, 256, 0, stream>>>(hlin, att_src1, att_dst1, asrc, adst);
  aggregate_kernel<<<NUM_GRAPHS, 256, 0, stream>>>(hlin, asrc, adst, src, ea,
      lin_edge1, att_edge1, bias1, hbuf, hbf, /*self_loop=*/0);

  // ---- conv2 ----  (tiles: 4096 M x 4 groups = 16384 waves = 2048 blocks)
  gemm_bf16_kernel<HC><<<2048, 256, 0, stream>>>(hbf, W2t, nullptr, hlin, NNODES, HC, 0);
  attn_kernel<<<2048, 256, 0, stream>>>(hlin, att_src2, att_dst2, asrc, adst);
  aggregate_kernel<<<NUM_GRAPHS, 256, 0, stream>>>(hlin, asrc, adst, src, ea,
      lin_edge2, att_edge2, bias2, hbuf, hbf, /*self_loop=*/1);

  // ---- conv3 ----
  gemm_bf16_kernel<HC><<<2048, 256, 0, stream>>>(hbf, W3t, nullptr, hlin, NNODES, HC, 0);
  attn_kernel<<<2048, 256, 0, stream>>>(hlin, att_src3, att_dst3, asrc, adst);
  aggregate_kernel<<<NUM_GRAPHS, 256, 0, stream>>>(hlin, asrc, adst, src, ea,
      lin_edge3, att_edge3, bias3, hbuf, hbf, /*self_loop=*/1);

  // ---- head ----  (fc1 tiles: 640 M x 2 groups = 1280 waves = 160 blocks)
  pool_comb_kernel<<<NUM_GRAPHS, 256, 0, stream>>>(hbuf, comb);
  gemm_bf16_kernel<FC1_IN><<<160, 256, 0, stream>>>(comb, fc1t, fc1_b, fc1o, NROWS, FC1_OUT, 1);
  fc2_kernel<<<(NROWS * ACT_DIM + 255) / 256, 256, 0, stream>>>(fc1o, fc2_w, fc2_b, log_std, out);
}
